// LineartoMel_real_26182120636748
// MI455X (gfx1250) — matrix operand order repaired and checked by reading, not compiled, not run on hardware
//
#include <hip/hip_runtime.h>
#include <hip/hip_bf16.h>
#include <stddef.h>

typedef __attribute__((ext_vector_type(16))) __bf16 v16bf;
typedef __attribute__((ext_vector_type(8)))  __bf16 v8bf;
typedef __attribute__((ext_vector_type(8)))  float  v8f;
typedef __attribute__((ext_vector_type(4)))  float  v4f;

constexpr int Bn    = 8;
constexpr int Fn    = 257;
constexpr int NCOMB = 10;
constexpr int Tn    = 1000;
constexpr int Hn    = 64;
constexpr int NCHn  = 4;
constexpr int NMELn = 40;
constexpr int Wk    = 3;
constexpr int TT    = 16;      // WMMA N (time tile)
constexpr int NRING = 48;      // 3 tiles of 16 columns
constexpr int RSTRIDE = 200;   // bf16 per ring column (192 used; 100 dwords -> conflict-free)
constexpr int XSTRIDE = 40;    // bf16 per xcol column (30 used; 20 dwords -> conflict-free)
constexpr int NT    = 63;      // tiles 0..62 produce output

// ---------------------------------------------------------------------------
// Fused 3-layer per-frequency conv, bf16 WMMA, im2col ring buffers over time.
// One block of 4 waves per (b,f); wave w owns M-chunk mc=w of layers 1+2,
// wave 3 owns layer 3.  grid = B*F.
// Ring layout: ring[col][K], K = 3c+kk, col=t holds h(c, t+kk-1) at elem 3c+kk.
// B fragment (K-chunk kc) for lane L: 16 consecutive K at col = colbase+L%16,
// K0 = kc*32 + 16*(L/16)  ->  two ds_load_b128.
// ---------------------------------------------------------------------------
__global__ __launch_bounds__(128) void l2mel_fused(
    const float* __restrict__ x,     // (B,F,NCOMB,T)
    const float* __restrict__ w1,    // (F,H,NCOMB,W)
    const float* __restrict__ b1,    // (F,H)
    const float* __restrict__ w2,    // (F,H,H,W)
    const float* __restrict__ b2,    // (F,H)
    const float* __restrict__ w3,    // (F,NCH,H,W)
    const float* __restrict__ b3,    // (F,NCH)
    const int*   __restrict__ seg,   // (F)
    float* __restrict__ wmel)        // (B,NMEL,NCH,T) accumulator (pre-zeroed)
{
    __shared__ __align__(32) __bf16 ldsA1[4 * 512];            //  4 KB
    __shared__ __align__(32) __bf16 ldsA2[24 * 512];           // 24 KB
    __shared__ __align__(32) __bf16 ldsA3[6 * 512];            //  6 KB
    __shared__ __align__(32) __bf16 h1ring[NRING * RSTRIDE];   // 18.75 KB
    __shared__ __align__(32) __bf16 h2ring[NRING * RSTRIDE];   // 18.75 KB
    __shared__ __align__(32) __bf16 xcol[TT * XSTRIDE];        // 1.25 KB
    __shared__ __align__(32) float lbias1[Hn];
    __shared__ __align__(32) float lbias2[Hn];
    __shared__ __align__(32) float lbias3[16];

    const int tid  = threadIdx.x;   // 0..127
    const int wv   = tid >> 5;      // wave 0..3
    const int lane = tid & 31;
    const int Lhi  = lane >> 4;     // 0/1
    const int Ln   = lane & 15;
    const int bid  = blockIdx.x;
    const int bb   = bid / Fn;
    const int ff   = bid - bb * Fn;
    const size_t xrowbase = ((size_t)bb * Fn + ff) * NCOMB * Tn;

    // ---- pack A fragments (weights -> bf16, WMMA A layout), split by wave ----
    // A element j of lane L maps to K = 16*Lhi + j, M = L%16.
    {
        const int mc = wv;                       // wave w packs its own M-chunk
        const int m  = mc * 16 + Ln;
        #pragma unroll
        for (int j = 0; j < 16; ++j) {
            const int K = Lhi * 16 + j;
            float v = 0.f;
            if (K < NCOMB * Wk) {
                const int c = K / 3, kk = K - 3 * c;
                v = w1[(((size_t)ff * Hn + m) * NCOMB + c) * Wk + kk];
            }
            ldsA1[mc * 512 + lane * 16 + j] = (__bf16)v;
        }
        for (int kc = 0; kc < 6; ++kc) {
            #pragma unroll
            for (int j = 0; j < 16; ++j) {
                const int Kf = Lhi * 16 + j;
                const int K  = kc * 32 + Kf;
                const int c = K / 3, kk = K - 3 * c;
                const float v = w2[(((size_t)ff * Hn + m) * Hn + c) * Wk + kk];
                ldsA2[(mc * 6 + kc) * 512 + lane * 16 + j] = (__bf16)v;
            }
        }
    }
    for (int kc = wv; kc < 6; kc += 4) {         // A3 frags split across waves
        #pragma unroll
        for (int j = 0; j < 16; ++j) {
            const int Kf = Lhi * 16 + j;
            const int K  = kc * 32 + Kf;
            const int c = K / 3, kk = K - 3 * c;
            float v = 0.f;
            if (Ln < NCHn)
                v = w3[(((size_t)ff * NCHn + Ln) * Hn + c) * Wk + kk];
            ldsA3[kc * 512 + lane * 16 + j] = (__bf16)v;
        }
    }
    // biases
    if (tid < Hn) {
        lbias1[tid] = b1[(size_t)ff * Hn + tid];
        lbias2[tid] = b2[(size_t)ff * Hn + tid];
    }
    if (tid < 16) lbias3[tid] = (tid < NCHn) ? b3[(size_t)ff * NCHn + tid] : 0.f;

    // zero rings (implements t<0 padding + never-written K slots) and xcol
    {
        unsigned* r1 = (unsigned*)h1ring;
        unsigned* r2 = (unsigned*)h2ring;
        for (int i = tid; i < (NRING * RSTRIDE) / 2; i += 128) { r1[i] = 0u; r2[i] = 0u; }
        unsigned* xz = (unsigned*)xcol;
        for (int i = tid; i < (TT * XSTRIDE) / 2; i += 128) xz[i] = 0u;
    }
    __syncthreads();

    const int segm = seg[ff];

    // helpers ---------------------------------------------------------------
    auto loadA = [&](const __bf16* base, int frag) -> v16bf {
        return *(const v16bf*)(base + frag * 512 + lane * 16);
    };
    auto gatherB = [&](const __bf16* ring, int kc, int colbase) -> v16bf {
        const __bf16* p = ring + (colbase + Ln) * RSTRIDE + kc * 32 + Lhi * 16;
        const v8bf lo = *(const v8bf*)p;
        const v8bf hi = *(const v8bf*)(p + 8);
        return __builtin_shufflevector(lo, hi, 0,1,2,3,4,5,6,7,8,9,10,11,12,13,14,15);
    };
    auto gatherB1 = [&]() -> v16bf {
        const __bf16* p = xcol + Ln * XSTRIDE + Lhi * 16;
        const v8bf lo = *(const v8bf*)p;
        const v8bf hi = *(const v8bf*)(p + 8);
        return __builtin_shufflevector(lo, hi, 0,1,2,3,4,5,6,7,8,9,10,11,12,13,14,15);
    };
    auto initAcc = [&](const float* lb, int mc) -> v8f {
        const int base = mc * 16 + 8 * Lhi;
        const v4f lo = *(const v4f*)&lb[base];
        const v4f hi = *(const v4f*)&lb[base + 4];
        return __builtin_shufflevector(lo, hi, 0,1,2,3,4,5,6,7);
    };
    // scatter relu(acc) into im2col ring: value (c,t) -> ring[t+1-kk][3c+kk]
    auto storeH = [&](__bf16* ring, int mc, int jt, v8f acc) {
        const int colbase = (jt % 3) * TT;
        const int tg = jt * TT + Ln;
        int col0 = colbase + Ln + 1;  if (col0 >= NRING) col0 -= NRING;   // kk=0
        int col1 = colbase + Ln;                                          // kk=1
        int col2 = colbase + Ln - 1;  if (col2 < 0) col2 += NRING;        // kk=2
        __bf16* p0 = ring + col0 * RSTRIDE + 0;
        __bf16* p1 = ring + col1 * RSTRIDE + 1;
        __bf16* p2 = ring + col2 * RSTRIDE + 2;
        const float zf = (tg < Tn) ? 1.f : 0.f;   // conv zero-padding past T
        #pragma unroll
        for (int v = 0; v < 8; ++v) {
            const int c3 = 3 * (mc * 16 + v + 8 * Lhi);
            const float r = fmaxf(acc[v], 0.f) * zf;
            const __bf16 h = (__bf16)r;
            p0[c3] = h;
            p1[c3] = h;
            p2[c3] = h;
        }
    };

    // software pipeline: h1 tile (step), h2 tile (step-1), output tile (step-2)
    for (int step = 0; step <= NT + 1; ++step) {
        const int j1 = step;
        // ---- stage x tile in im2col layout (all 128 threads) ----
        if (j1 <= NT - 1) {
            const int s = j1 * TT;
            for (int e = tid; e < NCOMB * 18; e += 128) {
                const int c = e / 18, tt = e - 18 * c;
                const int t = s - 1 + tt;
                float v = 0.f;
                if (t >= 0 && t < Tn) v = x[xrowbase + (size_t)c * Tn + t];
                const __bf16 h = (__bf16)v;
                #pragma unroll
                for (int kk = 0; kk < 3; ++kk) {
                    const int col = tt - kk;
                    if (col >= 0 && col < TT) xcol[col * XSTRIDE + 3 * c + kk] = h;
                }
            }
        }
        __syncthreads();
        // ---- layer 1: wave w computes M-chunk mc=w ----
        if (j1 <= NT - 1) {
            const v16bf bfx = gatherB1();
            const int mc = wv;
            v8f acc = initAcc(lbias1, mc);
            const v16bf af = loadA(ldsA1, mc);
            acc = __builtin_amdgcn_wmma_f32_16x16x32_bf16(
                    false, af, false, bfx, (short)0, acc, false, false);
            storeH(h1ring, mc, j1, acc);
        }
        __syncthreads();
        // ---- layer 2: wave w computes M-chunk mc=w ----
        const int j2 = step - 1;
        if (j2 >= 0 && j2 <= NT - 1) {
            const int colbase = (j2 % 3) * TT;
            v16bf bf2[6];
            #pragma unroll
            for (int kc = 0; kc < 6; ++kc) bf2[kc] = gatherB(h1ring, kc, colbase);
            const int mc = wv;
            v8f acc = initAcc(lbias2, mc);
            #pragma unroll
            for (int kc = 0; kc < 6; ++kc) {
                const v16bf af = loadA(ldsA2, mc * 6 + kc);
                acc = __builtin_amdgcn_wmma_f32_16x16x32_bf16(
                        false, af, false, bf2[kc], (short)0, acc, false, false);
            }
            storeH(h2ring, mc, j2, acc);
        }
        __syncthreads();
        // ---- layer 3 + mel segment-sum (wave 3 only) ----
        const int jo = step - 2;
        if (wv == 3 && jo >= 0 && jo <= NT - 1) {
            const int colbase = (jo % 3) * TT;
            v8f acc;
            {
                const v4f lo = *(const v4f*)&lbias3[8 * Lhi];
                const v4f hi = *(const v4f*)&lbias3[8 * Lhi + 4];
                acc = __builtin_shufflevector(lo, hi, 0,1,2,3,4,5,6,7);
            }
            v16bf bf3[6];
            #pragma unroll
            for (int kc = 0; kc < 6; ++kc) bf3[kc] = gatherB(h2ring, kc, colbase);
            #pragma unroll
            for (int kc = 0; kc < 6; ++kc) {
                const v16bf af = loadA(ldsA3, kc);
                acc = __builtin_amdgcn_wmma_f32_16x16x32_bf16(
                        false, af, false, bf3[kc], (short)0, acc, false, false);
            }
            if (Lhi == 0) {                       // rows 0..3 live in lanes 0..15
                const int t = jo * TT + Ln;
                if (t < Tn) {
                    #pragma unroll
                    for (int v = 0; v < NCHn; ++v)
                        atomicAdd(&wmel[(((size_t)bb * NMELn + segm) * NCHn + v) * Tn + t],
                                  acc[v]);
                }
            }
        }
    }
}

// ---------------------------------------------------------------------------
// lmfb = log1p(sum_ch mfb * W_mel)
// ---------------------------------------------------------------------------
__global__ void mel_finalize_k(const float* __restrict__ mfb,
                               const float* __restrict__ wmel,
                               float* __restrict__ out)
{
    const int i = blockIdx.x * blockDim.x + threadIdx.x;
    if (i >= Bn * NMELn * Tn) return;
    const int t  = i % Tn;
    const int bm = i / Tn;
    float s = 0.f;
    #pragma unroll
    for (int ch = 0; ch < NCHn; ++ch) {
        const size_t idx = ((size_t)bm * NCHn + ch) * Tn + t;
        s += mfb[idx] * wmel[idx];
    }
    out[i] = log1pf(s);
}

extern "C" void kernel_launch(void* const* d_in, const int* in_sizes, int n_in,
                              void* d_out, int out_size, void* d_ws, size_t ws_size,
                              hipStream_t stream)
{
    (void)in_sizes; (void)n_in; (void)out_size; (void)ws_size;
    const float* x   = (const float*)d_in[0];
    const float* mfb = (const float*)d_in[1];
    const float* w1  = (const float*)d_in[2];
    const float* b1  = (const float*)d_in[3];
    const float* w2  = (const float*)d_in[4];
    const float* b2  = (const float*)d_in[5];
    const float* w3  = (const float*)d_in[6];
    const float* b3  = (const float*)d_in[7];
    const int*   seg = (const int*)d_in[8];

    float* wmel = (float*)d_ws;                       // (B,NMEL,NCH,T) = 5.12 MB
    float* out  = (float*)d_out;

    hipMemsetAsync(wmel, 0, (size_t)Bn * NMELn * NCHn * Tn * sizeof(float), stream);

    l2mel_fused<<<Bn * Fn, 128, 0, stream>>>(x, w1, b1, w2, b2, w3, b3, seg, wmel);

    const int n2 = Bn * NMELn * Tn;
    mel_finalize_k<<<(n2 + 255) / 256, 256, 0, stream>>>(mfb, wmel, out);
}
